// RoPEALiBiLatentSelfAttention_20194936226121
// MI455X (gfx1250) — matrix operand order repaired and checked by reading, not compiled, not run on hardware
//
#include <hip/hip_runtime.h>
#include <math.h>
#include <stdint.h>

// Problem shape (fixed by the reference)
#define BB 8
#define LL 1024
#define DM 1024
#define PJ 512
#define HH 16
#define HD 64

typedef _Float16 h16;
typedef __attribute__((ext_vector_type(8)))  _Float16 v8h;
typedef __attribute__((ext_vector_type(16))) _Float16 v16h;
typedef __attribute__((ext_vector_type(8)))  float    v8f;
typedef __attribute__((ext_vector_type(4)))  unsigned u32x4;
typedef __attribute__((ext_vector_type(8)))  unsigned u32x8;

static __device__ __forceinline__ int laneid() { return (int)(threadIdx.x & 31); }

// A fragment: 16x32 f16 tile, row-major, leading dim ld (halves).
// ISA 7.12.2: lanes 0-15 -> M=lane, halves 0..7 = K 0..7, halves 8..15 = K 16..23
//             lanes 16-31 -> M=lane-16, K 8..15 and K 24..31.
static __device__ __forceinline__ v16h frag_a(const h16* tile, int ld) {
  const int l = laneid();
  const h16* p = tile + (l & 15) * ld + ((l >> 4) << 3);
  v8h lo = *(const v8h*)(p);
  v8h hi = *(const v8h*)(p + 16);
  v16h f;
#pragma unroll
  for (int i = 0; i < 8; ++i) { f[i] = lo[i]; f[i + 8] = hi[i]; }
  return f;
}

// B fragment: 32x16 tile given TRANSPOSED (Bt[n][k], leading dim ld).
// ISA: lane&15 = N column; lanes 0-15 K 0..7,16..23; lanes 16-31 K 8..15,24..31.
static __device__ __forceinline__ v16h frag_b(const h16* tileT, int ld) {
  const int l = laneid();
  const h16* p = tileT + (l & 15) * ld + ((l >> 4) << 3);
  v8h lo = *(const v8h*)(p);
  v8h hi = *(const v8h*)(p + 16);
  v16h f;
#pragma unroll
  for (int i = 0; i < 8; ++i) { f[i] = lo[i]; f[i + 8] = hi[i]; }
  return f;
}

static __device__ __forceinline__ v8f wmma_f16(v16h a, v16h b, v8f c) {
  return __builtin_amdgcn_wmma_f32_16x16x32_f16(false, a, false, b, (short)0, c,
                                                false, false);
}

// CDNA5 async copy: 16B global -> LDS, tracked by ASYNCcnt, no VGPR data path.
// lds_off = low 32 bits of a shared pointer (flat LDS aperture: addr[31:0]).
static __device__ __forceinline__ void async_ld16(unsigned lds_off, const void* g) {
  asm volatile("global_load_async_to_lds_b128 %0, %1, off"
               :: "v"(lds_off), "v"((unsigned long long)(uintptr_t)g)
               : "memory");
}
static __device__ __forceinline__ void wait_async_le4() {
  asm volatile("s_wait_asynccnt 0x4" ::: "memory");
}
static __device__ __forceinline__ void wait_async_0() {
  asm volatile("s_wait_asynccnt 0x0" ::: "memory");
}

// ---------------------------------------------------------------- converters
__global__ void cvt_f16_kernel(const float* __restrict__ x, h16* __restrict__ y,
                               long n) {
  long i = (long)blockIdx.x * blockDim.x + threadIdx.x;
  long stride = (long)gridDim.x * blockDim.x;
  for (; i < n; i += stride) y[i] = (h16)x[i];
}

// x: (R, C) f32 row-major  ->  y: (C, R) f16 row-major (transpose + convert)
__global__ void cvt_t_f16_kernel(const float* __restrict__ x, h16* __restrict__ y,
                                 int R, int C) {
  long i = (long)blockIdx.x * blockDim.x + threadIdx.x;
  if (i >= (long)R * C) return;
  int r = (int)(i / C), c = (int)(i % C);
  y[(long)c * R + r] = (h16)x[i];
}

// ---------------------------------------------------------------- WMMA GEMM
// C[M,N] = A[M,K](f16,rm) * Bt[N,K](f16,rm) + bias[N]
// 128x128 block, 8 waves (4M x 2N), each wave 32x64.
// Double-buffered LDS slabs filled with global_load_async_to_lds_b128;
// slab rows padded to 40 halves (80B) to rotate LDS banks.
#define GLDT 40
template <typename OutT>
__global__ void __launch_bounds__(256)
gemm_wmma_kernel(const h16* __restrict__ A, const h16* __restrict__ Bt,
                 const float* __restrict__ bias, OutT* __restrict__ C,
                 int M, int N, int K) {
  __shared__ __align__(32) h16 As[2 * 128 * GLDT];
  __shared__ __align__(32) h16 Bs[2 * 128 * GLDT];

  const int n0 = blockIdx.x * 128;
  const int m0 = blockIdx.y * 128;
  const int tid = threadIdx.x;
  const int wave = tid >> 5;
  const int wm = (wave & 3) * 32;
  const int wn = (wave >> 2) * 64;

  const v8f vzero = {0.f, 0.f, 0.f, 0.f, 0.f, 0.f, 0.f, 0.f};
  v8f acc[2][4];
#pragma unroll
  for (int i = 0; i < 2; ++i)
#pragma unroll
    for (int j = 0; j < 4; ++j) acc[i][j] = vzero;

  // staging: thread t owns 32B = row t/2, halves (t&1)*16 .. +16 of the slab
  const int ldr = tid >> 1;
  const int ldc = (tid & 1) * 16;
  const h16* gA = A + (long)(m0 + ldr) * K + ldc;
  const h16* gB = Bt + (long)(n0 + ldr) * K + ldc;
  const unsigned laBase =
      (unsigned)(uintptr_t)As + (unsigned)(ldr * GLDT + ldc) * 2u;
  const unsigned lbBase =
      (unsigned)(uintptr_t)Bs + (unsigned)(ldr * GLDT + ldc) * 2u;

  auto stage = [&](int k0, int buf) {
    const unsigned boff = (unsigned)(buf * 128 * GLDT * 2);
    async_ld16(laBase + boff,      gA + k0);
    async_ld16(laBase + boff + 16, gA + k0 + 8);
    async_ld16(lbBase + boff,      gB + k0);
    async_ld16(lbBase + boff + 16, gB + k0 + 8);
  };

  stage(0, 0);
  const int nk = K >> 5;
  for (int kc = 0; kc < nk; ++kc) {
    if (kc + 1 < nk) {
      stage((kc + 1) << 5, (kc + 1) & 1);  // prefetch next slab (other buffer)
      wait_async_le4();                    // slab kc done; next 4 in flight
    } else {
      wait_async_0();
    }
    __syncthreads();  // slab kc visible to all waves

    const h16* Ab = As + (kc & 1) * 128 * GLDT;
    const h16* Bb = Bs + (kc & 1) * 128 * GLDT;
    v16h bf[4];
#pragma unroll
    for (int tj = 0; tj < 4; ++tj) bf[tj] = frag_b(Bb + (wn + tj * 16) * GLDT, GLDT);
#pragma unroll
    for (int ti = 0; ti < 2; ++ti) {
      v16h af = frag_a(Ab + (wm + ti * 16) * GLDT, GLDT);
#pragma unroll
      for (int tj = 0; tj < 4; ++tj) acc[ti][tj] = wmma_f16(af, bf[tj], acc[ti][tj]);
    }
    __syncthreads();  // reads done before this buffer is re-staged
  }

  const int lane = tid & 31;
  const int cn = lane & 15;
  const int rb = (lane >> 4) * 8;
#pragma unroll
  for (int ti = 0; ti < 2; ++ti)
#pragma unroll
    for (int tj = 0; tj < 4; ++tj) {
      const int col = n0 + wn + tj * 16 + cn;
      const float bv = bias[col];
#pragma unroll
      for (int v = 0; v < 8; ++v) {
        const long row = m0 + wm + ti * 16 + rb + v;
        C[row * (long)N + col] = (OutT)(acc[ti][tj][v] + bv);
      }
    }
}

// ------------------------------------------------------- RoPE + head layout
__global__ void rope_layout_kernel(const h16* __restrict__ q,
                                   const h16* __restrict__ k,
                                   const h16* __restrict__ v,
                                   const float* __restrict__ pos,
                                   h16* __restrict__ Qh, h16* __restrict__ Kh,
                                   h16* __restrict__ Vt) {
  const long idx = (long)blockIdx.x * blockDim.x + threadIdx.x;
  if (idx >= (long)BB * LL * HH * (HD / 2)) return;
  const int j = (int)(idx & 31);
  const int h = (int)((idx >> 5) & 15);
  const int l = (int)((idx >> 9) & 1023);
  const int b = (int)(idx >> 19);

  const long src = ((long)(b * LL + l)) * DM + h * HD;
  const float f = pos[l * (HD / 2) + j];
  const float cs = __cosf(f), sn = __sinf(f);
  const float scaling = 0.125f;  // 64^-0.5, folded into Q

  const float q1 = (float)q[src + 2 * j], q2 = (float)q[src + 2 * j + 1];
  const float k1 = (float)k[src + 2 * j], k2 = (float)k[src + 2 * j + 1];
  const float v1 = (float)v[src + 2 * j], v2 = (float)v[src + 2 * j + 1];

  const long dqk = (((long)(b * HH + h)) * LL + l) * HD;
  Qh[dqk + 2 * j]     = (h16)((q1 * cs - q2 * sn) * scaling);
  Qh[dqk + 2 * j + 1] = (h16)((q1 * sn + q2 * cs) * scaling);
  Kh[dqk + 2 * j]     = (h16)(k1 * cs - k2 * sn);
  Kh[dqk + 2 * j + 1] = (h16)(k1 * sn + k2 * cs);

  const long dv = ((long)(b * HH + h)) * HD * LL;
  Vt[dv + (long)(2 * j) * LL + l]     = (h16)v1;
  Vt[dv + (long)(2 * j + 1) * LL + l] = (h16)v2;
}

// ---------------------------------------------------------------- attention
// One workgroup = (b,h) x 32 query rows; 32x1024 f32 score strip in LDS.
// Q strip is DMA'd by the Tensor Data Mover (pad 64->72-half rows).
#define QS_LD 72
#define P_LD  1032
#define ATT_SMEM_S   (32 * 1024 * 4)
#define ATT_SMEM_P   (32 * P_LD * 2)
#define ATT_SMEM_Q   (32 * QS_LD * 2)
#define ATT_SMEM_RED (32 * 8 * 4)
#define ATT_SMEM_ROW (64 * 4)
#define ATT_SMEM_TOTAL \
  (ATT_SMEM_S + ATT_SMEM_P + ATT_SMEM_Q + ATT_SMEM_RED + ATT_SMEM_ROW)

__global__ void __launch_bounds__(256)
attn_kernel(const h16* __restrict__ Qh, const h16* __restrict__ Kh,
            const h16* __restrict__ Vt, const float* __restrict__ alibi,
            const uint8_t* __restrict__ mask, h16* __restrict__ Oh) {
  extern __shared__ __align__(32) char smem[];
  float* Sbuf  = (float*)smem;                               // [32][1024]
  h16*   Pbuf  = (h16*)(smem + ATT_SMEM_S);                  // [32][P_LD]
  h16*   Qs    = (h16*)(smem + ATT_SMEM_S + ATT_SMEM_P);     // [32][QS_LD]
  float* red   = (float*)(smem + ATT_SMEM_S + ATT_SMEM_P + ATT_SMEM_Q);
  float* rowst = red + 32 * 8;                               // [64]

  const int bh = blockIdx.x;   // 0..127
  const int qb = blockIdx.y;   // 0..31
  const int b = bh >> 4, h = bh & 15;
  const int q0 = qb * 32;
  const int tid = threadIdx.x;
  const int wave = tid >> 5;
  const int lane = tid & 31;

  // ---- TDM: wave 0 DMA-loads the 32x64 Q strip (row-padded to 72 halves)
  const long qbase = ((long)bh * LL + q0) * HD;
  if (wave == 0) {
    const unsigned long long ga = (unsigned long long)(uintptr_t)(Qh + qbase);
    const unsigned qs_lds = (unsigned)(uintptr_t)Qs;
    // D# group0: count=1, lds_addr, global_addr[56:0], type=2
    u32x4 g0 = {1u, qs_lds, (unsigned)ga,
                (unsigned)((ga >> 32) & 0x01FFFFFFu) | (2u << 30)};
    // D# group1: data_size=2B, pad_enable, pad every 32 DW by 4 DW,
    // tensor_dim0=64, tensor_dim1=32, tile 64x32, dim0_stride=64
    u32x8 g1 = {(1u << 16) | (1u << 20) | (4u << 22) | (3u << 25),
                64u << 16,   // tensor_dim0[15:0] << 16
                32u << 16,   // tensor_dim1[15:0] << 16
                64u << 16,   // tile_dim0 << 16
                32u,         // tile_dim1
                64u,         // tensor_dim0_stride[31:0]
                0u, 0u};
    asm volatile("tensor_load_to_lds %0, %1" :: "s"(g0), "s"(g1) : "memory");
    __builtin_amdgcn_s_wait_tensorcnt(0);
  }
  __syncthreads();

  // ---- phase 1: S = Q K^T (32 x 1024); Q fragments hoisted (invariant)
  v16h afq[2][2];
#pragma unroll
  for (int mi = 0; mi < 2; ++mi)
#pragma unroll
    for (int kk = 0; kk < 2; ++kk)
      afq[mi][kk] = frag_a(Qs + (mi * 16) * QS_LD + kk * 32, QS_LD);

  for (int ct = wave; ct < 64; ct += 8) {
    const h16* kt = Kh + ((long)bh * LL + ct * 16) * HD;  // Bt[n][k] rows
    v16h bf0 = frag_b(kt, HD);
    v16h bf1 = frag_b(kt + 32, HD);
    const int cn = lane & 15, rb = (lane >> 4) * 8;
#pragma unroll
    for (int mi = 0; mi < 2; ++mi) {
      v8f acc = {0.f, 0.f, 0.f, 0.f, 0.f, 0.f, 0.f, 0.f};
      acc = wmma_f16(afq[mi][0], bf0, acc);
      acc = wmma_f16(afq[mi][1], bf1, acc);
#pragma unroll
      for (int v = 0; v < 8; ++v)
        Sbuf[(mi * 16 + rb + v) * 1024 + ct * 16 + cn] = acc[v];
    }
  }
  __syncthreads();

  // ---- phase 2: +alibi, mask, softmax (8 threads per row, 128 cols each)
  const int r = tid >> 3;
  const int cp = tid & 7;
  float* srow = Sbuf + r * 1024;
  const float* arow = alibi + ((long)h * LL + (q0 + r)) * LL;
  const uint8_t* mrow = mask + (long)b * LL;

  float mx = -INFINITY;
  for (int c = cp * 128; c < cp * 128 + 128; ++c) {
    float s = srow[c] + arow[c];   // score scale folded into Q
    if (mrow[c]) s = -INFINITY;
    srow[c] = s;
    mx = fmaxf(mx, s);
  }
  red[r * 8 + cp] = mx;
  __syncthreads();
  if (tid < 32) {
    float m2 = red[tid * 8];
#pragma unroll
    for (int i = 1; i < 8; ++i) m2 = fmaxf(m2, red[tid * 8 + i]);
    rowst[tid] = m2;
  }
  __syncthreads();
  const float rmax = rowst[r];
  float sum = 0.f;
  for (int c = cp * 128; c < cp * 128 + 128; ++c) {
    float e = __expf(srow[c] - rmax);
    srow[c] = e;
    sum += e;
  }
  __syncthreads();
  red[r * 8 + cp] = sum;
  __syncthreads();
  if (tid < 32) {
    float s2 = 0.f;
#pragma unroll
    for (int i = 0; i < 8; ++i) s2 += red[tid * 8 + i];
    rowst[32 + tid] = 1.0f / s2;
  }
  __syncthreads();
  const float rinv = rowst[32 + r];
  for (int c = cp * 128; c < cp * 128 + 128; ++c)
    Pbuf[r * P_LD + c] = (h16)(srow[c] * rinv);
  __syncthreads();

  // ---- phase 3: O = P V (32 x 64), one 16x16 tile per wave, K=1024
  {
    const int mi = wave & 1, nj = wave >> 1;
    const h16* vt = Vt + ((long)bh * HD + nj * 16) * LL;  // Vt[d][k] rows
    v8f acc = {0.f, 0.f, 0.f, 0.f, 0.f, 0.f, 0.f, 0.f};
#pragma unroll 4
    for (int k0 = 0; k0 < 1024; k0 += 32)
      acc = wmma_f16(frag_a(Pbuf + (mi * 16) * P_LD + k0, P_LD),
                     frag_b(vt + k0, 1024), acc);
    const int cn = lane & 15, rb = (lane >> 4) * 8;
#pragma unroll
    for (int v = 0; v < 8; ++v) {
      const long row = (long)b * LL + q0 + mi * 16 + rb + v;  // (B,L,D)
      Oh[row * DM + h * HD + nj * 16 + cn] = (h16)acc[v];
    }
  }
}

// ---------------------------------------------------------------- launcher
extern "C" void kernel_launch(void* const* d_in, const int* in_sizes, int n_in,
                              void* d_out, int out_size, void* d_ws,
                              size_t ws_size, hipStream_t stream) {
  const float*   src   = (const float*)d_in[0];
  const float*   alibi = (const float*)d_in[1];
  const float*   pos   = (const float*)d_in[2];
  const uint8_t* mask  = (const uint8_t*)d_in[3];
  const float*   Wl    = (const float*)d_in[4];
  const float*   bl    = (const float*)d_in[5];
  const float*   Wq    = (const float*)d_in[6];
  const float*   bq    = (const float*)d_in[7];
  const float*   Wk    = (const float*)d_in[8];
  const float*   bk    = (const float*)d_in[9];
  const float*   Wv    = (const float*)d_in[10];
  const float*   bv    = (const float*)d_in[11];
  const float*   Wo    = (const float*)d_in[12];
  const float*   bo    = (const float*)d_in[13];
  float* out = (float*)d_out;

  char* p = (char*)d_ws;
  h16* srcH  = (h16*)p; p += (long)BB * LL * DM * 2;
  h16* WlT   = (h16*)p; p += (long)DM * PJ * 2;
  h16* WqT   = (h16*)p; p += (long)PJ * DM * 2;
  h16* WkT   = (h16*)p; p += (long)PJ * DM * 2;
  h16* WvT   = (h16*)p; p += (long)PJ * DM * 2;
  h16* WoT   = (h16*)p; p += (long)DM * DM * 2;
  h16* projH = (h16*)p; p += (long)BB * LL * PJ * 2;
  h16* qH    = (h16*)p; p += (long)BB * LL * DM * 2;
  h16* kH    = (h16*)p; p += (long)BB * LL * DM * 2;
  h16* vH    = (h16*)p; p += (long)BB * LL * DM * 2;
  h16* QhH   = (h16*)p; p += (long)BB * LL * DM * 2;   // (B,H,L,HD)
  h16* KhH   = (h16*)p; p += (long)BB * LL * DM * 2;   // (B,H,L,HD)
  h16* VtH   = (h16*)p; p += (long)BB * LL * DM * 2;   // (B,H,HD,L)
  h16* OH    = (h16*)p; p += (long)BB * LL * DM * 2;   // attn out (B,L,D)

  const int M = BB * LL;  // 8192

  cvt_f16_kernel<<<4096, 256, 0, stream>>>(src, srcH, (long)M * DM);
  cvt_t_f16_kernel<<<(DM * PJ + 255) / 256, 256, 0, stream>>>(Wl, WlT, DM, PJ);
  cvt_t_f16_kernel<<<(PJ * DM + 255) / 256, 256, 0, stream>>>(Wq, WqT, PJ, DM);
  cvt_t_f16_kernel<<<(PJ * DM + 255) / 256, 256, 0, stream>>>(Wk, WkT, PJ, DM);
  cvt_t_f16_kernel<<<(PJ * DM + 255) / 256, 256, 0, stream>>>(Wv, WvT, PJ, DM);
  cvt_t_f16_kernel<<<(DM * DM + 255) / 256, 256, 0, stream>>>(Wo, WoT, DM, DM);

  gemm_wmma_kernel<h16><<<dim3(PJ / 128, M / 128), 256, 0, stream>>>(
      srcH, WlT, bl, projH, M, PJ, DM);

  gemm_wmma_kernel<h16><<<dim3(DM / 128, M / 128), 256, 0, stream>>>(
      projH, WqT, bq, qH, M, DM, PJ);
  gemm_wmma_kernel<h16><<<dim3(DM / 128, M / 128), 256, 0, stream>>>(
      projH, WkT, bk, kH, M, DM, PJ);
  gemm_wmma_kernel<h16><<<dim3(DM / 128, M / 128), 256, 0, stream>>>(
      projH, WvT, bv, vH, M, DM, PJ);

  rope_layout_kernel<<<(BB * LL * HH * (HD / 2)) / 256, 256, 0, stream>>>(
      qH, kH, vH, pos, QhH, KhH, VtH);

  attn_kernel<<<dim3(BB * HH, LL / 32), 256, ATT_SMEM_TOTAL, stream>>>(
      QhH, KhH, VtH, alibi, mask, OH);

  gemm_wmma_kernel<float><<<dim3(DM / 128, M / 128), 256, 0, stream>>>(
      OH, WoT, bo, out, M, DM, DM);
}
